// MoE_73675868996049
// MI455X (gfx1250) — matrix operand order repaired and checked by reading, not compiled, not run on hardware
//
#include <hip/hip_runtime.h>

// ---------------- types ----------------
typedef unsigned short u16_t;
typedef u16_t  u16x8_t  __attribute__((ext_vector_type(8)));
typedef u16_t  u16x16_t __attribute__((ext_vector_type(16)));
typedef unsigned int u32x4_t __attribute__((ext_vector_type(4)));
typedef int    i32x4_t  __attribute__((ext_vector_type(4)));
typedef float  f32x4_t  __attribute__((ext_vector_type(4)));
typedef float  f32x8_t  __attribute__((ext_vector_type(8)));
typedef __bf16 bfx16_t  __attribute__((ext_vector_type(16)));

#define DIM      1024
#define N_EXP    16
#define TOPK     4
#define T_TOK    2048
#define INTER    512
#define SH_INTER 2048

#define BM 128
#define BN 64
#define KSTEP 64      // LDS K-tile: two WMMA K-steps per barrier pair

// ---------------- CDNA5 async global->LDS path (guarded) ----------------
#if __has_builtin(__builtin_amdgcn_global_load_async_to_lds_b128)
#define ASYNC_OK 1
#else
#define ASYNC_OK 0
#endif

// builtin expects: (v4i addrspace(1)* src, v4i addrspace(3)* dst, imm offset, cpol)
typedef __attribute__((address_space(1))) i32x4_t gvec_t;
typedef __attribute__((address_space(3))) i32x4_t lvec_t;

static __device__ __forceinline__ gvec_t* as_gbl(const void* p) {
    return (gvec_t*)(unsigned long long)p;
}
static __device__ __forceinline__ lvec_t* as_lds(const void* p) {
    // flat LDS address truncated to 32 bits == LDS byte offset (ISA 10.2 aperture rule)
    return (lvec_t*)(unsigned int)(unsigned long long)p;
}

#if ASYNC_OK
#if __has_builtin(__builtin_amdgcn_s_wait_asynccnt)
#define WAIT_ASYNC(n) __builtin_amdgcn_s_wait_asynccnt(n)
#else
#define WAIT_ASYNC(n) asm volatile("s_wait_asynccnt %0" :: "i"(n) : "memory")
#endif
#else
#define WAIT_ASYNC(n) do {} while (0)
#endif

static __device__ __forceinline__ u16_t f32_to_bf16(float f) {
    union { float f; unsigned u; } v; v.f = f;
    unsigned u = v.u;
    u += 0x7FFFu + ((u >> 16) & 1u);   // round-to-nearest-even
    return (u16_t)(u >> 16);
}

static __device__ __forceinline__ bfx16_t pack16(u16x8_t lo, u16x8_t hi) {
    u16x16_t t;
#pragma unroll
    for (int i = 0; i < 8; i++) { t[i] = lo[i]; t[i + 8] = hi[i]; }
    return __builtin_bit_cast(bfx16_t, t);
}

static __device__ __forceinline__ float silu(float x) {
    return x / (1.0f + __expf(-x));
}

// ---------------- small prep kernels ----------------
__global__ void k_zero_cnt(int* cnt) {
    if (threadIdx.x < N_EXP) cnt[threadIdx.x] = 0;
}

// elementwise fp32 -> bf16, 8 elements / thread
__global__ void k_cvt(const float* __restrict__ s, u16_t* __restrict__ d, int n) {
    int i = (blockIdx.x * 256 + threadIdx.x) * 8;
    if (i >= n) return;
    f32x4_t f0 = *(const f32x4_t*)(s + i);
    f32x4_t f1 = *(const f32x4_t*)(s + i + 4);
    u16x8_t o;
#pragma unroll
    for (int j = 0; j < 4; j++) { o[j] = f32_to_bf16(f0[j]); o[j + 4] = f32_to_bf16(f1[j]); }
    *(u16x8_t*)(d + i) = o;
}

// src f32 [K][N] row-major  ->  dst bf16 [N][K] row-major (per blockIdx.z matrix)
__global__ __launch_bounds__(256) void k_tr(const float* __restrict__ src, u16_t* __restrict__ dst,
                                            int K, int N, size_t sStride, size_t dStride) {
    const float* s = src + (size_t)blockIdx.z * sStride;
    u16_t*       d = dst + (size_t)blockIdx.z * dStride;
    __shared__ float tile[32][33];
    int bk = blockIdx.x * 32, bn = blockIdx.y * 32;
    int tx = threadIdx.x & 31, ty = threadIdx.x >> 5;   // ty 0..7
#pragma unroll
    for (int i = 0; i < 4; i++) {
        int r = ty + i * 8;
        tile[r][tx] = s[(size_t)(bk + r) * N + bn + tx];
    }
    __syncthreads();
#pragma unroll
    for (int i = 0; i < 4; i++) {
        int r = ty + i * 8;
        d[(size_t)(bn + r) * K + bk + tx] = f32_to_bf16(tile[tx][r]);
    }
}

// ---------------- router ----------------
// one wave per token; lane e<16 computes logit for expert e (fp32)
__global__ __launch_bounds__(128) void k_gate(const float* __restrict__ x, const float* __restrict__ gw,
                                              int* __restrict__ cnt, int* __restrict__ tok,
                                              float* __restrict__ wt) {
    int wid  = threadIdx.x >> 5;
    int t    = blockIdx.x * 4 + wid;
    int lane = threadIdx.x & 31;
    float logit = 0.f;
    if (lane < N_EXP) {
        const float* xr = x  + (size_t)t * DIM;
        const float* gr = gw + (size_t)lane * DIM;
#pragma unroll 4
        for (int d = 0; d < DIM; d++) logit += xr[d] * gr[d];
    }
    float sc[N_EXP];
#pragma unroll
    for (int e = 0; e < N_EXP; e++) sc[e] = __shfl(logit, e, 32);
    if (lane != 0) return;
    // softmax
    float m = sc[0];
    for (int e = 1; e < N_EXP; e++) m = fmaxf(m, sc[e]);
    float s = 0.f;
    for (int e = 0; e < N_EXP; e++) { sc[e] = __expf(sc[e] - m); s += sc[e]; }
    float inv = 1.f / s;
    for (int e = 0; e < N_EXP; e++) sc[e] *= inv;
    // group-limited routing: 4 groups of 4, keep top-2 groups by group max
    float gs[4];
    for (int g = 0; g < 4; g++) {
        float mg = sc[g * 4];
        for (int j = 1; j < 4; j++) mg = fmaxf(mg, sc[g * 4 + j]);
        gs[g] = mg;
    }
    int g0 = 0; for (int g = 1; g < 4; g++) if (gs[g] > gs[g0]) g0 = g;
    int g1 = -1; for (int g = 0; g < 4; g++) { if (g == g0) continue; if (g1 < 0 || gs[g] > gs[g1]) g1 = g; }
    float msk[N_EXP];
    for (int e = 0; e < N_EXP; e++) {
        int g = e >> 2;
        msk[e] = (g == g0 || g == g1) ? sc[e] : -1.f;
    }
    for (int k = 0; k < TOPK; k++) {
        int be = 0;
        for (int e = 1; e < N_EXP; e++) if (msk[e] > msk[be]) be = e;
        float w = sc[be];           // ROUTE_SCALE = 1.0, no renorm
        msk[be] = -2.f;
        int p = atomicAdd(&cnt[be], 1);
        tok[be * T_TOK + p] = t;
        wt [be * T_TOK + p] = w;
    }
}

// ---------------- SwiGLU up GEMM: Out = bf16( silu(A@B1) * (A@B3) [* route_w] ) ----------------
// A: bf16 [rows][Kdim] (optionally gathered by token list), B1/B3: bf16 [N][Kdim] (transposed)
__global__ __launch_bounds__(256) void k_up(const u16_t* __restrict__ Xb,
                                            const u16_t* __restrict__ B1base,
                                            const u16_t* __restrict__ B3base, size_t bStrideE,
                                            int Kdim, int Nld,
                                            u16_t* __restrict__ OutBase, size_t outStrideE,
                                            const int* __restrict__ tokBase,
                                            const float* __restrict__ wtBase,
                                            const int* __restrict__ cntBase, int gated) {
    int e  = blockIdx.z;
    int tb = blockIdx.x * BM;
    int nb = blockIdx.y * BN;
    int cntv = 1 << 30;
    const int*   tokL = nullptr;
    const float* wtL  = nullptr;
    if (gated) {
        cntv = cntBase[e];
        if (tb >= cntv) return;
        tokL = tokBase + (size_t)e * T_TOK;
        wtL  = wtBase  + (size_t)e * T_TOK;
    }
    const u16_t* B1  = B1base + (size_t)e * bStrideE;
    const u16_t* B3  = B3base + (size_t)e * bStrideE;
    u16_t*       Out = OutBase + (size_t)e * outStrideE;

    __shared__ u16_t As [2][BM * KSTEP];   // 2 x 16 KB
    __shared__ u16_t Bs1[2][BN * KSTEP];   // 2 x  8 KB
    __shared__ u16_t Bs3[2][BN * KSTEP];   // 2 x  8 KB

    int t = threadIdx.x;
    // A-tile copy map: 2 threads/row, 32 u16 (4x16B) each
    int aRow = t >> 1;
    int aCol = (t & 1) * 32;
    int aTok;
    if (gated) { int p = tb + aRow; aTok = (p < cntv) ? tokL[p] : 0; }
    else       { aTok = tb + aRow; }
    const u16_t* aSrc = Xb + (size_t)aTok * (size_t)Kdim + aCol;
    // B-tile copy map: 4 threads/row, 16 u16 (2x16B) each, per matrix
    int bRow = t >> 2;
    int bCol = (t & 3) * 16;
    const u16_t* b1Src = B1 + (size_t)(nb + bRow) * Kdim + bCol;
    const u16_t* b3Src = B3 + (size_t)(nb + bRow) * Kdim + bCol;

    auto issue = [&](int p, int kk) {
#if ASYNC_OK
#pragma unroll
        for (int c = 0; c < 4; c++)
            __builtin_amdgcn_global_load_async_to_lds_b128(
                as_gbl(aSrc + kk + c * 8),
                as_lds(&As[p][aRow * KSTEP + aCol + c * 8]), 0, 0);
#pragma unroll
        for (int c = 0; c < 2; c++) {
            __builtin_amdgcn_global_load_async_to_lds_b128(
                as_gbl(b1Src + kk + c * 8),
                as_lds(&Bs1[p][bRow * KSTEP + bCol + c * 8]), 0, 0);
            __builtin_amdgcn_global_load_async_to_lds_b128(
                as_gbl(b3Src + kk + c * 8),
                as_lds(&Bs3[p][bRow * KSTEP + bCol + c * 8]), 0, 0);
        }
#else
        u32x4_t a[4], r1[2], r3[2];
#pragma unroll
        for (int c = 0; c < 4; c++) a[c] = *(const u32x4_t*)(aSrc + kk + c * 8);
#pragma unroll
        for (int c = 0; c < 2; c++) {
            r1[c] = *(const u32x4_t*)(b1Src + kk + c * 8);
            r3[c] = *(const u32x4_t*)(b3Src + kk + c * 8);
        }
#pragma unroll
        for (int c = 0; c < 4; c++) *(u32x4_t*)&As[p][aRow * KSTEP + aCol + c * 8] = a[c];
#pragma unroll
        for (int c = 0; c < 2; c++) {
            *(u32x4_t*)&Bs1[p][bRow * KSTEP + bCol + c * 8] = r1[c];
            *(u32x4_t*)&Bs3[p][bRow * KSTEP + bCol + c * 8] = r3[c];
        }
#endif
    };

    int lane = t & 31, wid = t >> 5;
    int wm = (wid & 3) * 32;   // wave M offset within CTA tile
    int wn = (wid >> 2) * 32;  // wave N offset
    int lm = lane & 15, lg = lane >> 4;

    f32x8_t acc1[2][2] = {};
    f32x8_t acc3[2][2] = {};

    issue(0, 0);
    int pb = 0;
    for (int kk = 0; kk < Kdim; kk += KSTEP) {
        if (kk + KSTEP < Kdim) { issue(pb ^ 1, kk + KSTEP); WAIT_ASYNC(8); }
        else                   { WAIT_ASYNC(0); }
        __syncthreads();

#pragma unroll
        for (int ks = 0; ks < KSTEP; ks += 32) {
            bfx16_t af[2], b1f[2], b3f[2];
#pragma unroll
            for (int mf = 0; mf < 2; mf++) {
                const u16_t* ap = &As[pb][(wm + mf * 16 + lm) * KSTEP + ks];
                af[mf] = pack16(*(const u16x8_t*)(ap + lg * 16),
                                *(const u16x8_t*)(ap + lg * 16 + 8));
            }
#pragma unroll
            for (int nf = 0; nf < 2; nf++) {
                const u16_t* p1 = &Bs1[pb][(wn + nf * 16 + lm) * KSTEP + ks + lg * 16];
                b1f[nf] = pack16(*(const u16x8_t*)p1, *(const u16x8_t*)(p1 + 8));
                const u16_t* p3 = &Bs3[pb][(wn + nf * 16 + lm) * KSTEP + ks + lg * 16];
                b3f[nf] = pack16(*(const u16x8_t*)p3, *(const u16x8_t*)(p3 + 8));
            }
#pragma unroll
            for (int mf = 0; mf < 2; mf++)
#pragma unroll
                for (int nf = 0; nf < 2; nf++) {
                    acc1[mf][nf] = __builtin_amdgcn_wmma_f32_16x16x32_bf16(
                        false, af[mf], false, b1f[nf], (short)0, acc1[mf][nf], false, false);
                    acc3[mf][nf] = __builtin_amdgcn_wmma_f32_16x16x32_bf16(
                        false, af[mf], false, b3f[nf], (short)0, acc3[mf][nf], false, false);
                }
        }
        __syncthreads();
        pb ^= 1;
    }

    // epilogue: D layout -> row = base + r + 8*lg, col = base + lm
    if (gated) {
#pragma unroll
        for (int mf = 0; mf < 2; mf++) {
            int rowBase = tb + wm + mf * 16 + 8 * lg;
            float scale[8];
#pragma unroll
            for (int r = 0; r < 8; r++) {
                int p = rowBase + r;
                scale[r] = (p < cntv) ? wtL[p] : 0.f;
            }
#pragma unroll
            for (int nf = 0; nf < 2; nf++) {
                int col = nb + wn + nf * 16 + lm;
#pragma unroll
                for (int r = 0; r < 8; r++) {
                    float a = silu(acc1[mf][nf][r]) * acc3[mf][nf][r] * scale[r];
                    Out[(size_t)(rowBase + r) * Nld + col] = f32_to_bf16(a);
                }
            }
        }
    } else {
#pragma unroll
        for (int mf = 0; mf < 2; mf++) {
            int rowBase = tb + wm + mf * 16 + 8 * lg;
#pragma unroll
            for (int nf = 0; nf < 2; nf++) {
                int col = nb + wn + nf * 16 + lm;
#pragma unroll
                for (int r = 0; r < 8; r++) {
                    float a = silu(acc1[mf][nf][r]) * acc3[mf][nf][r];
                    Out[(size_t)(rowBase + r) * Nld + col] = f32_to_bf16(a);
                }
            }
        }
    }
}

// ---------------- down GEMM: Out(+)= A @ B ----------------
__global__ __launch_bounds__(256) void k_down(const u16_t* __restrict__ Abase, size_t aStrideE, int Kdim,
                                              const u16_t* __restrict__ Bbase, size_t bStrideE,
                                              float* __restrict__ Out,
                                              const int* __restrict__ tokBase,
                                              const int* __restrict__ cntBase, int gated) {
    int e  = blockIdx.z;
    int tb = blockIdx.x * BM;
    int nb = blockIdx.y * BN;
    int cntv = 1 << 30;
    const int* tokL = nullptr;
    if (gated) {
        cntv = cntBase[e];
        if (tb >= cntv) return;
        tokL = tokBase + (size_t)e * T_TOK;
    }
    const u16_t* A = Abase + (size_t)e * aStrideE;
    const u16_t* B = Bbase + (size_t)e * bStrideE;

    __shared__ u16_t As[2][BM * KSTEP];   // 2 x 16 KB
    __shared__ u16_t Bs[2][BN * KSTEP];   // 2 x  8 KB

    int t = threadIdx.x;
    int aRow = t >> 1;
    int aCol = (t & 1) * 32;
    const u16_t* aSrc = A + (size_t)(tb + aRow) * Kdim + aCol;
    int bRow = t >> 2;
    int bCol = (t & 3) * 16;
    const u16_t* bSrc = B + (size_t)(nb + bRow) * Kdim + bCol;

    auto issue = [&](int p, int kk) {
#if ASYNC_OK
#pragma unroll
        for (int c = 0; c < 4; c++)
            __builtin_amdgcn_global_load_async_to_lds_b128(
                as_gbl(aSrc + kk + c * 8),
                as_lds(&As[p][aRow * KSTEP + aCol + c * 8]), 0, 0);
#pragma unroll
        for (int c = 0; c < 2; c++)
            __builtin_amdgcn_global_load_async_to_lds_b128(
                as_gbl(bSrc + kk + c * 8),
                as_lds(&Bs[p][bRow * KSTEP + bCol + c * 8]), 0, 0);
#else
        u32x4_t a[4], b[2];
#pragma unroll
        for (int c = 0; c < 4; c++) a[c] = *(const u32x4_t*)(aSrc + kk + c * 8);
#pragma unroll
        for (int c = 0; c < 2; c++) b[c] = *(const u32x4_t*)(bSrc + kk + c * 8);
#pragma unroll
        for (int c = 0; c < 4; c++) *(u32x4_t*)&As[p][aRow * KSTEP + aCol + c * 8] = a[c];
#pragma unroll
        for (int c = 0; c < 2; c++) *(u32x4_t*)&Bs[p][bRow * KSTEP + bCol + c * 8] = b[c];
#endif
    };

    int lane = t & 31, wid = t >> 5;
    int wm = (wid & 3) * 32;
    int wn = (wid >> 2) * 32;
    int lm = lane & 15, lg = lane >> 4;

    f32x8_t acc[2][2] = {};

    issue(0, 0);
    int pb = 0;
    for (int kk = 0; kk < Kdim; kk += KSTEP) {
        if (kk + KSTEP < Kdim) { issue(pb ^ 1, kk + KSTEP); WAIT_ASYNC(6); }
        else                   { WAIT_ASYNC(0); }
        __syncthreads();

#pragma unroll
        for (int ks = 0; ks < KSTEP; ks += 32) {
            bfx16_t af[2], bf[2];
#pragma unroll
            for (int mf = 0; mf < 2; mf++) {
                const u16_t* ap = &As[pb][(wm + mf * 16 + lm) * KSTEP + ks];
                af[mf] = pack16(*(const u16x8_t*)(ap + lg * 16),
                                *(const u16x8_t*)(ap + lg * 16 + 8));
            }
#pragma unroll
            for (int nf = 0; nf < 2; nf++) {
                const u16_t* bp = &Bs[pb][(wn + nf * 16 + lm) * KSTEP + ks + lg * 16];
                bf[nf] = pack16(*(const u16x8_t*)bp, *(const u16x8_t*)(bp + 8));
            }
#pragma unroll
            for (int mf = 0; mf < 2; mf++)
#pragma unroll
                for (int nf = 0; nf < 2; nf++)
                    acc[mf][nf] = __builtin_amdgcn_wmma_f32_16x16x32_bf16(
                        false, af[mf], false, bf[nf], (short)0, acc[mf][nf], false, false);
        }
        __syncthreads();
        pb ^= 1;
    }

    if (gated) {
#pragma unroll
        for (int mf = 0; mf < 2; mf++) {
            int rowBase = tb + wm + mf * 16 + 8 * lg;
#pragma unroll
            for (int nf = 0; nf < 2; nf++) {
                int col = nb + wn + nf * 16 + lm;
#pragma unroll
                for (int r = 0; r < 8; r++) {
                    int p = rowBase + r;
                    if (p < cntv)
                        atomicAdd(&Out[(size_t)tokL[p] * DIM + col], acc[mf][nf][r]);
                }
            }
        }
    } else {
#pragma unroll
        for (int mf = 0; mf < 2; mf++) {
            int rowBase = tb + wm + mf * 16 + 8 * lg;
#pragma unroll
            for (int nf = 0; nf < 2; nf++) {
                int col = nb + wn + nf * 16 + lm;
#pragma unroll
                for (int r = 0; r < 8; r++)
                    Out[(size_t)(rowBase + r) * DIM + col] = acc[mf][nf][r];
            }
        }
    }
}

// ---------------- host launch ----------------
extern "C" void kernel_launch(void* const* d_in, const int* in_sizes, int n_in,
                              void* d_out, int out_size, void* d_ws, size_t ws_size,
                              hipStream_t stream) {
    const float* x   = (const float*)d_in[0];
    const float* gw  = (const float*)d_in[1];
    const float* w1  = (const float*)d_in[2];
    const float* w2  = (const float*)d_in[3];
    const float* w3  = (const float*)d_in[4];
    const float* ws1 = (const float*)d_in[5];
    const float* ws2 = (const float*)d_in[6];
    const float* ws3 = (const float*)d_in[7];
    float* out = (float*)d_out;
    char*  ws  = (char*)d_ws;

    // workspace layout (bytes)
    u16_t* xb   = (u16_t*)(ws + 0x00000000);  // 2048*1024  bf16  (4 MB)
    u16_t* w1t  = (u16_t*)(ws + 0x00400000);  // 16*[512][1024]   (16 MB)
    u16_t* w3t  = (u16_t*)(ws + 0x01400000);  // 16 MB
    u16_t* w2t  = (u16_t*)(ws + 0x02400000);  // 16*[1024][512]   (16 MB)
    u16_t* ws1t = (u16_t*)(ws + 0x03400000);  // [2048][1024]     (4 MB)
    u16_t* ws3t = (u16_t*)(ws + 0x03800000);  // 4 MB
    u16_t* ws2t = (u16_t*)(ws + 0x03C00000);  // [1024][2048]     (4 MB)
    u16_t* H    = (u16_t*)(ws + 0x04000000);  // [2048][2048] bf16 (8 MB)
    u16_t* Abuf = (u16_t*)(ws + 0x04800000);  // 16*[2048][512] bf16 (32 MB)
    int*   cnt  = (int*)  (ws + 0x06800000);  // 16 ints
    int*   tok  = (int*)  (ws + 0x06810000);  // 16*2048 ints
    float* wt   = (float*)(ws + 0x06830000);  // 16*2048 floats

    (void)in_sizes; (void)n_in; (void)out_size; (void)ws_size;

    k_zero_cnt<<<1, 32, 0, stream>>>(cnt);

    // x -> bf16
    k_cvt<<<(T_TOK * DIM / 8 + 255) / 256, 256, 0, stream>>>(x, xb, T_TOK * DIM);

    // weight transpose+convert:  [K][N] f32 -> [N][K] bf16
    k_tr<<<dim3(DIM / 32, INTER / 32, N_EXP), 256, 0, stream>>>(w1, w1t, DIM, INTER,
        (size_t)DIM * INTER, (size_t)DIM * INTER);
    k_tr<<<dim3(DIM / 32, INTER / 32, N_EXP), 256, 0, stream>>>(w3, w3t, DIM, INTER,
        (size_t)DIM * INTER, (size_t)DIM * INTER);
    k_tr<<<dim3(INTER / 32, DIM / 32, N_EXP), 256, 0, stream>>>(w2, w2t, INTER, DIM,
        (size_t)INTER * DIM, (size_t)INTER * DIM);
    k_tr<<<dim3(DIM / 32, SH_INTER / 32, 1), 256, 0, stream>>>(ws1, ws1t, DIM, SH_INTER, 0, 0);
    k_tr<<<dim3(DIM / 32, SH_INTER / 32, 1), 256, 0, stream>>>(ws3, ws3t, DIM, SH_INTER, 0, 0);
    k_tr<<<dim3(SH_INTER / 32, DIM / 32, 1), 256, 0, stream>>>(ws2, ws2t, SH_INTER, DIM, 0, 0);

    // routing (fp32)
    k_gate<<<T_TOK / 4, 128, 0, stream>>>(x, gw, cnt, tok, wt);

    // shared expert: H = silu(x@ws1)*(x@ws3);  out = H@ws2   (writes full output)
    k_up<<<dim3(T_TOK / BM, SH_INTER / BN, 1), 256, 0, stream>>>(
        xb, ws1t, ws3t, 0, DIM, SH_INTER, H, 0, nullptr, nullptr, nullptr, 0);
    k_down<<<dim3(T_TOK / BM, DIM / BN, 1), 256, 0, stream>>>(
        H, 0, SH_INTER, ws2t, 0, out, nullptr, nullptr, 0);

    // routed experts: gather -> SwiGLU*route_w -> scatter-add
    k_up<<<dim3(T_TOK / BM, INTER / BN, N_EXP), 256, 0, stream>>>(
        xb, w1t, w3t, (size_t)INTER * DIM, DIM, INTER,
        Abuf, (size_t)T_TOK * INTER, tok, wt, cnt, 1);
    k_down<<<dim3(T_TOK / BM, DIM / BN, N_EXP), 256, 0, stream>>>(
        Abuf, (size_t)T_TOK * INTER, INTER, w2t, (size_t)DIM * INTER, out, tok, cnt, 1);
}
